// W8DWLinear_41850161332883
// MI455X (gfx1250) — 8-bit matrix operand order repaired and checked by reading, compiled, not run on hardware
//
#include <hip/hip_runtime.h>

#define D_IN   4096
#define D_OUT  4096
#define ROWS   8192
#define KOUT   16
#define Q2_INV (1.0f / 16129.0f)

typedef __attribute__((ext_vector_type(8))) int v8i;

__global__ __launch_bounds__(256) void prep_x_kernel(
    const float* __restrict__ x, const int* __restrict__ idx_unq,
    signed char* __restrict__ xq, float* __restrict__ xmax,
    float* __restrict__ xunq) {
  __shared__ int   sidx[KOUT];
  __shared__ float smax[8];
  const int m   = blockIdx.x;
  const int tid = threadIdx.x;
  if (tid < KOUT) sidx[tid] = idx_unq[tid];
  __syncthreads();

  const float* row = x + (size_t)m * D_IN;
  float vals[16];
  float vmax = 0.0f;
#pragma unroll
  for (int i = 0; i < 16; ++i) {
    const int c = tid + i * 256;
    float v = row[c];
    bool outl = false;
#pragma unroll
    for (int j = 0; j < KOUT; ++j) outl |= (c == sidx[j]);
    v = outl ? 0.0f : v;
    vals[i] = v;
    vmax = fmaxf(vmax, fabsf(v));
  }
#pragma unroll
  for (int off = 16; off > 0; off >>= 1)
    vmax = fmaxf(vmax, __shfl_xor(vmax, off, 32));
  if ((tid & 31) == 0) smax[tid >> 5] = vmax;
  __syncthreads();
  if (tid == 0) {
    float r = smax[0];
#pragma unroll
    for (int w = 1; w < 8; ++w) r = fmaxf(r, smax[w]);
    smax[0] = r;
    xmax[m] = r;
  }
  __syncthreads();
  const float scale = 127.0f / smax[0];

  signed char* qrow = xq + (size_t)m * D_IN;
#pragma unroll
  for (int i = 0; i < 16; ++i) {
    const int c = tid + i * 256;
    qrow[c] = (signed char)(int)rintf(vals[i] * scale);
  }
  if (tid < KOUT) xunq[(size_t)m * KOUT + tid] = row[sidx[tid]];
}

__global__ __launch_bounds__(256) void prep_w_kernel(
    const float* __restrict__ wf, signed char* __restrict__ wq) {
  const size_t i = ((size_t)blockIdx.x * 256 + threadIdx.x) * 4;
  float4 v = *(const float4*)(wf + i);
  const int b0 = ((int)rintf(v.x)) & 0xff;
  const int b1 = ((int)rintf(v.y)) & 0xff;
  const int b2 = ((int)rintf(v.z)) & 0xff;
  const int b3 = ((int)rintf(v.w)) & 0xff;
  *(int*)(wq + i) = b0 | (b1 << 8) | (b2 << 16) | (b3 << 24);
}

__device__ inline v8i load_a_frag(const signed char* base, int off) {
  int4 d0 = *(const int4*)(base + off);
  int4 d1 = *(const int4*)(base + off + 32);
  v8i a;
  a[0] = d0.x; a[1] = d0.y; a[2] = d0.z; a[3] = d0.w;
  a[4] = d1.x; a[5] = d1.y; a[6] = d1.z; a[7] = d1.w;
  return a;
}

__device__ inline v8i load_b_frag(const signed char* base, int off) {
  int4 d0 = *(const int4*)(base + off);
  int4 d1 = *(const int4*)(base + off + 32);
  v8i r;
  r[0] = d0.x; r[1] = d0.y; r[2] = d0.z; r[3] = d0.w;
  r[4] = d1.x; r[5] = d1.y; r[6] = d1.z; r[7] = d1.w;
  return r;
}

__global__ __launch_bounds__(256) void gemm_iu8_kernel(
    const signed char* __restrict__ xq, const signed char* __restrict__ wq,
    const float* __restrict__ xmax, const float* __restrict__ wmax,
    const float* __restrict__ xunq, const float* __restrict__ wunq,
    const float* __restrict__ bias, float* __restrict__ out) {
  const int lane  = threadIdx.x & 31;
  const int lrow  = lane & 15;
  const int half  = lane >> 4;
  const int wave  = threadIdx.x >> 5;
  const int waveM = wave & 1;
  const int waveN = wave >> 1;
  const int mBase = blockIdx.y * 128 + waveM * 64;
  const int nBase = blockIdx.x * 256 + waveN * 64;

  v8i acc[4][4];
#pragma unroll
  for (int mi = 0; mi < 4; ++mi)
#pragma unroll
    for (int ni = 0; ni < 4; ++ni) acc[mi][ni] = {};

  int aoff[4], boff[4];
#pragma unroll
  for (int mi = 0; mi < 4; ++mi)
    aoff[mi] = (mBase + mi * 16 + lrow) * D_IN + half * 16;
#pragma unroll
  for (int ni = 0; ni < 4; ++ni)
    boff[ni] = (nBase + ni * 16 + lrow) * D_IN + half * 16;

  v8i af0[4], bf0[4], af1[4], bf1[4];

#pragma unroll
  for (int mi = 0; mi < 4; ++mi) af0[mi] = load_a_frag(xq, aoff[mi]);
#pragma unroll
  for (int ni = 0; ni < 4; ++ni) bf0[ni] = load_b_frag(wq, boff[ni]);

  int k0 = 0;
  for (; k0 < D_IN - 128; k0 += 128) {
    __builtin_prefetch(xq + aoff[0] + k0 + 128, 0, 3);
    __builtin_prefetch(wq + boff[0] + k0 + 128, 0, 3);
#pragma unroll
    for (int mi = 0; mi < 4; ++mi) af1[mi] = load_a_frag(xq, aoff[mi] + k0 + 64);
#pragma unroll
    for (int ni = 0; ni < 4; ++ni) bf1[ni] = load_b_frag(wq, boff[ni] + k0 + 64);
#pragma unroll
    for (int mi = 0; mi < 4; ++mi)
#pragma unroll
      for (int ni = 0; ni < 4; ++ni)
        acc[mi][ni] = __builtin_amdgcn_wmma_i32_16x16x64_iu8(
            true, af0[mi], true, bf0[ni], acc[mi][ni], false, false);
#pragma unroll
    for (int mi = 0; mi < 4; ++mi) af0[mi] = load_a_frag(xq, aoff[mi] + k0 + 128);
#pragma unroll
    for (int ni = 0; ni < 4; ++ni) bf0[ni] = load_b_frag(wq, boff[ni] + k0 + 128);
#pragma unroll
    for (int mi = 0; mi < 4; ++mi)
#pragma unroll
      for (int ni = 0; ni < 4; ++ni)
        acc[mi][ni] = __builtin_amdgcn_wmma_i32_16x16x64_iu8(
            true, af1[mi], true, bf1[ni], acc[mi][ni], false, false);
  }
#pragma unroll
  for (int mi = 0; mi < 4; ++mi) af1[mi] = load_a_frag(xq, aoff[mi] + k0 + 64);
#pragma unroll
  for (int ni = 0; ni < 4; ++ni) bf1[ni] = load_b_frag(wq, boff[ni] + k0 + 64);
#pragma unroll
  for (int mi = 0; mi < 4; ++mi)
#pragma unroll
    for (int ni = 0; ni < 4; ++ni)
      acc[mi][ni] = __builtin_amdgcn_wmma_i32_16x16x64_iu8(
          true, af0[mi], true, bf0[ni], acc[mi][ni], false, false);
#pragma unroll
  for (int mi = 0; mi < 4; ++mi)
#pragma unroll
    for (int ni = 0; ni < 4; ++ni)
      acc[mi][ni] = __builtin_amdgcn_wmma_i32_16x16x64_iu8(
          true, af1[mi], true, bf1[ni], acc[mi][ni], false, false);

  float wmx[4], bn2[4];
  float4 w0[4], w1[4], w2[4], w3[4];
#pragma unroll
  for (int ni = 0; ni < 4; ++ni) {
    const int n = nBase + ni * 16 + lrow;
    wmx[ni] = wmax[n];
    bn2[ni] = bias[n];
    const float4* wu = (const float4*)(wunq + (size_t)n * KOUT);
    w0[ni] = wu[0]; w1[ni] = wu[1]; w2[ni] = wu[2]; w3[ni] = wu[3];
  }

#pragma unroll
  for (int mi = 0; mi < 4; ++mi) {
#pragma unroll
    for (int r = 0; r < 8; ++r) {
      const int m = mBase + mi * 16 + half * 8 + r;
      const float xm = xmax[m];
      const float4* xu = (const float4*)(xunq + (size_t)m * KOUT);
      float4 a0 = xu[0], a1 = xu[1], a2 = xu[2], a3 = xu[3];
#pragma unroll
      for (int ni = 0; ni < 4; ++ni) {
        const int n = nBase + ni * 16 + lrow;
        float v = (float)acc[mi][ni][r] * Q2_INV;
        v = (float)(_Float16)v;
        v *= xm * wmx[ni];
        float dot = bn2[ni]
          + a0.x * w0[ni].x + a0.y * w0[ni].y + a0.z * w0[ni].z + a0.w * w0[ni].w
          + a1.x * w1[ni].x + a1.y * w1[ni].y + a1.z * w1[ni].z + a1.w * w1[ni].w
          + a2.x * w2[ni].x + a2.y * w2[ni].y + a2.z * w2[ni].z + a2.w * w2[ni].w
          + a3.x * w3[ni].x + a3.y * w3[ni].y + a3.z * w3[ni].z + a3.w * w3[ni].w;
        out[(size_t)m * D_OUT + n] = v + dot;
      }
    }
  }
}

extern "C" void kernel_launch(void* const* d_in, const int* in_sizes, int n_in,
                              void* d_out, int out_size, void* d_ws,
                              size_t ws_size, hipStream_t stream) {
  (void)in_sizes; (void)n_in; (void)out_size; (void)ws_size;
  const float* x          = (const float*)d_in[0];
  const float* weight_q   = (const float*)d_in[1];
  const float* w_max      = (const float*)d_in[2];
  const float* weight_unq = (const float*)d_in[3];
  const float* bias       = (const float*)d_in[4];
  const int*   idx_unq    = (const int*)d_in[5];
  float* out = (float*)d_out;

  char* ws = (char*)d_ws;
  signed char* xq   = (signed char*)ws;
  signed char* wq   = (signed char*)(ws + (size_t)ROWS * D_IN);
  float*       xmax = (float*)(ws + (size_t)ROWS * D_IN + (size_t)D_OUT * D_IN);
  float*       xunq = (float*)((char*)xmax + (size_t)ROWS * sizeof(float));

  prep_x_kernel<<<ROWS, 256, 0, stream>>>(x, idx_unq, xq, xmax, xunq);
  prep_w_kernel<<<(D_OUT * (size_t)D_IN) / 1024, 256, 0, stream>>>(weight_q, wq);

  dim3 grid(D_OUT / 256, ROWS / 128);
  gemm_iu8_kernel<<<grid, 256, 0, stream>>>(xq, wq, xmax, w_max, xunq,
                                            weight_unq, bias, out);
}
